// GIN_75127567942136
// MI455X (gfx1250) — matrix operand order repaired and checked by reading, not compiled, not run on hardware
//
#include <hip/hip_runtime.h>

#define DFEAT 128

typedef __attribute__((ext_vector_type(16))) __bf16 v16bf;
typedef __attribute__((ext_vector_type(8)))  float  v8f;

union BFrag {
    v16bf v;
    uint4 q[2];
};

__device__ __forceinline__ unsigned short f2bf(float f) {
    unsigned int u = __float_as_uint(f);
    unsigned int r = u + 0x7FFFu + ((u >> 16) & 1u);   // round-to-nearest-even
    return (unsigned short)(r >> 16);
}

// ---------------- u32 fill ----------------
__global__ void gin_fillu(unsigned* __restrict__ p, unsigned v, int n) {
    int i = blockIdx.x * blockDim.x + threadIdx.x;
    if (i < n) p[i] = v;
}

// ---------------- in-degree count (int atomics: 4B/edge, not 512B) ----------------
__global__ void gin_degree(const long long* __restrict__ dst, unsigned* __restrict__ deg, int E) {
    int e = blockIdx.x * blockDim.x + threadIdx.x;
    if (e < E) atomicAdd(&deg[(int)dst[e]], 1u);
}

// ---------------- exclusive prefix sum over N (single block, LDS Hillis-Steele) ----------------
__global__ void gin_scan(const unsigned* __restrict__ deg, unsigned* __restrict__ rowstart, int N) {
    __shared__ unsigned smem[1024];
    __shared__ unsigned carry;
    const int t = threadIdx.x;
    if (t == 0) carry = 0;
    __syncthreads();
    for (int base = 0; base < N; base += 1024) {
        int i = base + t;
        unsigned v = (i < N) ? deg[i] : 0u;
        smem[t] = v;
        __syncthreads();
        for (int off = 1; off < 1024; off <<= 1) {
            unsigned add = (t >= off) ? smem[t - off] : 0u;
            __syncthreads();
            smem[t] += add;
            __syncthreads();
        }
        if (i < N) rowstart[i] = carry + smem[t] - v;   // exclusive
        __syncthreads();
        if (t == 0) carry += smem[1023];
        __syncthreads();
    }
    if (t == 0) rowstart[N] = carry;
}

// ---------------- bucket edges into CSR order ----------------
__global__ void gin_bucket(const long long* __restrict__ src, const long long* __restrict__ dst,
                           unsigned* __restrict__ cursor, int* __restrict__ esrc, int E) {
    int e = blockIdx.x * blockDim.x + threadIdx.x;
    if (e < E) {
        int d = (int)dst[e];
        unsigned pos = atomicAdd(&cursor[d], 1u);       // cursor pre-seeded with rowstart
        esrc[pos] = (int)src[e];
    }
}

// ---------------- weights fp32 -> bf16 ----------------
__global__ void gin_cvt_bf16(const float* __restrict__ in, unsigned short* __restrict__ out, int n4) {
    int i = blockIdx.x * blockDim.x + threadIdx.x;
    if (i >= n4) return;
    const float4 v = *(const float4*)(in + i * 4);
    ushort4 o;
    o.x = f2bf(v.x); o.y = f2bf(v.y); o.z = f2bf(v.z); o.w = f2bf(v.w);
    *(ushort4*)(out + i * 4) = o;
}

// ---------------- fused gather-aggregate + GIN add + bf16 convert ----------------
// One wave per node; 32 lanes x 4 floats cover d=128. Register accumulation, no atomics,
// no sum buffer. 4-way unrolled gather -> 4 independent 512B loads in flight per wave.
__global__ void gin_aggregate_h(const float* __restrict__ x, const int* __restrict__ esrc,
                                const unsigned* __restrict__ rowstart,
                                unsigned short* __restrict__ hbf, int N) {
    int tid  = blockIdx.x * blockDim.x + threadIdx.x;
    int node = tid >> 5;
    int lane = tid & 31;
    if (node >= N) return;
    unsigned beg = rowstart[node];
    unsigned end = rowstart[node + 1];
    float4 acc = make_float4(0.f, 0.f, 0.f, 0.f);

    unsigned j = beg;
    for (; j + 4 <= end; j += 4) {
        int s0 = esrc[j + 0];
        int s1 = esrc[j + 1];
        int s2 = esrc[j + 2];
        int s3 = esrc[j + 3];
        const float4 v0 = *(const float4*)(x + (size_t)s0 * DFEAT + lane * 4);
        const float4 v1 = *(const float4*)(x + (size_t)s1 * DFEAT + lane * 4);
        const float4 v2 = *(const float4*)(x + (size_t)s2 * DFEAT + lane * 4);
        const float4 v3 = *(const float4*)(x + (size_t)s3 * DFEAT + lane * 4);
        acc.x += v0.x + v1.x + v2.x + v3.x;
        acc.y += v0.y + v1.y + v2.y + v3.y;
        acc.z += v0.z + v1.z + v2.z + v3.z;
        acc.w += v0.w + v1.w + v2.w + v3.w;
    }
    for (; j < end; ++j) {
        int s = esrc[j];
        const float4 v = *(const float4*)(x + (size_t)s * DFEAT + lane * 4);
        acc.x += v.x; acc.y += v.y; acc.z += v.z; acc.w += v.w;
    }

    float inv = 1.0f / fmaxf((float)(end - beg), 1.0f);
    const float4 xv = *(const float4*)(x + (size_t)node * DFEAT + lane * 4);
    ushort4 o;
    o.x = f2bf(xv.x + acc.x * inv);
    o.y = f2bf(xv.y + acc.y * inv);
    o.z = f2bf(xv.z + acc.z * inv);
    o.w = f2bf(xv.w + acc.w * inv);
    *(ushort4*)(hbf + (size_t)node * DFEAT + lane * 4) = o;
}

// ---------------- GEMM: out[N,128] = H[N,128] @ W^T + b, optional relu ----------------
// W (bf16, 32KB) staged in LDS once per block, shared by 8 waves. One wave per 16-row
// tile; A frags preloaded for all 4 K-steps; B fragments double-buffered from LDS so
// the next ds_load_b128 pair issues before the current WMMA retires.
template <int RELU>
__global__ void gin_gemm128(const unsigned short* __restrict__ H,
                            const unsigned short* __restrict__ Wb,
                            const float* __restrict__ bias,
                            float* __restrict__ out, int N) {
    __shared__ unsigned short lw[DFEAT * DFEAT];        // 32KB of 320KB/WGP

    // cooperative stage: 2048 uint4s / 256 threads = 8 x b128 each
    for (int i = threadIdx.x; i < (DFEAT * DFEAT) / 8; i += blockDim.x)
        ((uint4*)lw)[i] = ((const uint4*)Wb)[i];
    __syncthreads();

    const int wid  = (blockIdx.x * blockDim.x + threadIdx.x) >> 5;   // wave-uniform
    const int lane = threadIdx.x & 31;
    const int idx  = lane & 15;    // A: row-in-tile ; B: out-col-in-tile
    const int half = lane >> 4;

    int arow = wid * 16 + idx;
    if (arow >= N) arow = N - 1;   // clamp loads (tail / overshoot waves); stores guarded

    // Preload A fragments for all 4 K-steps (16x32 bf16 each):
    // lane half h holds K = h*16..h*16+15
    BFrag a[4];
#pragma unroll
    for (int kt = 0; kt < 4; ++kt) {
        const unsigned short* ap = H + (size_t)arow * DFEAT + kt * 32;
        a[kt].q[0] = *(const uint4*)(ap + half * 16);
        a[kt].q[1] = *(const uint4*)(ap + half * 16 + 8);
    }

    // B fragment address: 32x16 tile, B[k,o] = W[o,k]; lane holds K = half*16..+15 of W row o
    auto bptr = [&](int kt, int nt) -> const unsigned short* {
        return lw + (nt * 16 + idx) * DFEAT + kt * 32 + half * 16;
    };

    v8f acc[8] = {};
    BFrag b[2];
    {
        const unsigned short* bp = bptr(0, 0);
        b[0].q[0] = *(const uint4*)(bp);
        b[0].q[1] = *(const uint4*)(bp + 8);
    }
#pragma unroll
    for (int i = 0; i < 32; ++i) {                      // i = kt*8 + nt
        const int kt  = i >> 3;
        const int nt  = i & 7;
        const int cur = i & 1;
        if (i < 31) {                                   // prefetch next B fragment
            const int ni = i + 1;
            const unsigned short* bp = bptr(ni >> 3, ni & 7);
            b[cur ^ 1].q[0] = *(const uint4*)(bp);
            b[cur ^ 1].q[1] = *(const uint4*)(bp + 8);
        }
        acc[nt] = __builtin_amdgcn_wmma_f32_16x16x32_bf16(
            false, a[kt].v, false, b[cur].v, (short)0, acc[nt], false, false);
    }

    // D layout: VGPR v, lane: m = half*8 + v, n = idx
#pragma unroll
    for (int nt = 0; nt < 8; ++nt) {
        int col = nt * 16 + idx;
        float bv = bias[col];
#pragma unroll
        for (int v = 0; v < 8; ++v) {
            int r = wid * 16 + half * 8 + v;
            if (r < N) {
                float val = acc[nt][v] + bv;
                if (RELU) val = fmaxf(val, 0.f);
                out[(size_t)r * DFEAT + col] = val;
            }
        }
    }
}

extern "C" void kernel_launch(void* const* d_in, const int* in_sizes, int n_in,
                              void* d_out, int out_size, void* d_ws, size_t ws_size,
                              hipStream_t stream) {
    const float*     features = (const float*)d_in[0];
    const long long* src      = (const long long*)d_in[1];
    const long long* dst      = (const long long*)d_in[2];
    const float*     W1       = (const float*)d_in[3];
    const float*     b1       = (const float*)d_in[4];
    const float*     W2       = (const float*)d_in[5];
    const float*     b2       = (const float*)d_in[6];
    float*           out      = (float*)d_out;

    const int N = in_sizes[0] / DFEAT;
    const int E = in_sizes[1];

    // workspace carve-out (256B aligned)
    char*  ws  = (char*)d_ws;
    size_t off = 0;
    auto carve = [&](size_t bytes) -> char* {
        char* p = ws + off;
        off = (off + bytes + 255) & ~(size_t)255;
        return p;
    };
    unsigned*       deg      = (unsigned*)carve((size_t)N * 4);
    unsigned*       rowstart = (unsigned*)carve((size_t)(N + 1) * 4);
    unsigned*       cursor   = (unsigned*)carve((size_t)(N + 1) * 4);
    int*            esrc     = (int*)carve((size_t)E * 4);
    float*          x1       = (float*)carve((size_t)N * DFEAT * 4);
    unsigned short* hbf      = (unsigned short*)carve((size_t)N * DFEAT * 2);
    unsigned short* wbf      = (unsigned short*)carve((size_t)DFEAT * DFEAT * 2);

    const int TB = 256;
    const int n_w4       = (DFEAT * DFEAT) / 4;
    const int rowTiles   = (N + 15) / 16;
    const int gemmBlocks = (rowTiles + 7) / 8;          // 8 waves / 256-thread block
    const int aggBlocks  = (int)(((size_t)N * 32 + TB - 1) / TB);

    // ---- CSR build (once; graph shared by both layers) ----
    gin_fillu<<<(N + TB - 1) / TB, TB, 0, stream>>>(deg, 0u, N);
    gin_degree<<<(E + TB - 1) / TB, TB, 0, stream>>>(dst, deg, E);
    gin_scan<<<1, 1024, 0, stream>>>(deg, rowstart, N);
    hipMemcpyAsync(cursor, rowstart, (size_t)(N + 1) * 4, hipMemcpyDeviceToDevice, stream);
    gin_bucket<<<(E + TB - 1) / TB, TB, 0, stream>>>(src, dst, cursor, esrc, E);

    // ---- layer 1 ----
    gin_cvt_bf16<<<(n_w4 + TB - 1) / TB, TB, 0, stream>>>(W1, wbf, n_w4);
    gin_aggregate_h<<<aggBlocks, TB, 0, stream>>>(features, esrc, rowstart, hbf, N);
    gin_gemm128<1><<<gemmBlocks, TB, 0, stream>>>(hbf, wbf, b1, x1, N);

    // ---- layer 2 ----
    gin_cvt_bf16<<<(n_w4 + TB - 1) / TB, TB, 0, stream>>>(W2, wbf, n_w4);
    gin_aggregate_h<<<aggBlocks, TB, 0, stream>>>(x1, esrc, rowstart, hbf, N);
    gin_gemm128<0><<<gemmBlocks, TB, 0, stream>>>(hbf, wbf, b2, out, N);
}
